// TrajGRU_cell_25718264168982
// MI455X (gfx1250) — matrix operand order repaired and checked by reading, not compiled, not run on hardware
//
#include <hip/hip_runtime.h>
#include <hip/hip_bf16.h>
#include <math.h>

// ---------------- problem constants ----------------
#define Bn   4
#define Tn   10
#define CIN  8
#define Hn   96
#define Wn   96
#define HW   (Hn*Wn)          // 9216
#define HID  64
#define Ln   13
#define KK   (Ln*HID)         // 832  (wrap GEMM K)
#define NN   (3*HID)          // 192  (wrap GEMM N)
#define MM   (Bn*HW)          // 36864 (wrap GEMM M)
#define FCH  32               // flow-feature channels
#define FL2  (2*Ln)           // 26 flow channels

typedef __bf16 bf16_t;
typedef __attribute__((ext_vector_type(16))) __bf16 v16bf;
typedef __attribute__((ext_vector_type(8)))  __bf16 v8bf;
typedef __attribute__((ext_vector_type(8)))  float  v8f;

__device__ __forceinline__ float sigmoidf(float x) { return 1.0f / (1.0f + __expf(-x)); }

union frag16 { v16bf v; v8bf h[2]; };

// ---------------- weight fp32 -> bf16 (wrap_w, K-contiguous rows) ----------------
__global__ void k_cvt_bf16(const float* __restrict__ src, bf16_t* __restrict__ dst, int n) {
    int i = blockIdx.x * blockDim.x + threadIdx.x;
    if (i < n) dst[i] = (bf16_t)src[i];
}

// ---------------- h2f_w (32,64,5,5) -> [tap][oc][ic] bf16 ----------------
__global__ void k_prep_h2f(const float* __restrict__ w, bf16_t* __restrict__ out) {
    int i = blockIdx.x * blockDim.x + threadIdx.x;      // 25*32*64
    if (i >= 25 * 32 * 64) return;
    int ic  = i % 64;
    int oc  = (i / 64) % 32;
    int tap = i / (64 * 32);
    out[i] = (bf16_t)w[((size_t)oc * 64 + ic) * 25 + tap];
}

// ---------------- flows_w (26,32,5,5) -> [tap][oc(32 pad)][ic] bf16 ----------------
__global__ void k_prep_flows(const float* __restrict__ w, bf16_t* __restrict__ out) {
    int i = blockIdx.x * blockDim.x + threadIdx.x;      // 25*32*32
    if (i >= 25 * 32 * 32) return;
    int ic  = i % 32;
    int oc  = (i / 32) % 32;
    int tap = i / (32 * 32);
    out[i] = (oc < FL2) ? (bf16_t)w[((size_t)oc * 32 + ic) * 25 + tap] : (bf16_t)0.0f;
}

// ---------------- state (B,HID,H,W) fp32 -> hbf (B,HW,HID) bf16 NHWC ----------------
__global__ void k_cvt_nhwc(const float* __restrict__ src, bf16_t* __restrict__ dst) {
    int i = blockIdx.x * blockDim.x + threadIdx.x;      // B*HID*HW
    if (i >= Bn * HID * HW) return;
    int p = i % HW;
    int c = (i / HW) % HID;
    int b = i / (HW * HID);
    dst[((size_t)b * HW + p) * HID + c] = (bf16_t)src[((size_t)(b * HID + c)) * HW + p];
}

// ---------------- i2h: 3x3 conv, CIN -> 192, pad 1 (per timestep, direct) ----------------
__global__ void k_conv3x3_i2h(const float* __restrict__ in, const float* __restrict__ w,
                              const float* __restrict__ bias, float* __restrict__ out, int t) {
    int idx = blockIdx.x * blockDim.x + threadIdx.x;
    if (idx >= Bn * NN * HW) return;
    int p  = idx % HW;
    int oc = (idx / HW) % NN;
    int b  = idx / (HW * NN);
    int y = p / Wn, x = p % Wn;
    const float* xin = in + ((size_t)(b * Tn + t) * CIN) * HW;
    float sum = bias[oc];
    #pragma unroll
    for (int ic = 0; ic < CIN; ++ic) {
        const float* ip = xin + (size_t)ic * HW;
        const float* wp = w + ((size_t)oc * CIN + ic) * 9;
        #pragma unroll
        for (int ky = 0; ky < 3; ++ky) {
            int yy = y + ky - 1;
            if (yy < 0 || yy >= Hn) continue;
            #pragma unroll
            for (int kx = 0; kx < 3; ++kx) {
                int xx = x + kx - 1;
                if (xx < 0 || xx >= Wn) continue;
                sum = __builtin_fmaf(ip[yy * Wn + xx], wp[ky * 3 + kx], sum);
            }
        }
    }
    out[((size_t)(b * NN + oc)) * HW + p] = sum;
}

// ---------------- i2f pre-pass: fpre(NHWC) = bi + bh + conv5(x_t, i2f_w) ----------------
__global__ void k_i2f(const float* __restrict__ in, const float* __restrict__ wi,
                      const float* __restrict__ bi, const float* __restrict__ bh,
                      float* __restrict__ out, int t) {           // out: (B,HW,32)
    int idx = blockIdx.x * blockDim.x + threadIdx.x;
    if (idx >= Bn * FCH * HW) return;
    int p  = idx % HW;
    int oc = (idx / HW) % FCH;
    int b  = idx / (HW * FCH);
    int y = p / Wn, x = p % Wn;
    float sum = bi[oc] + bh[oc];
    const float* xin = in + ((size_t)(b * Tn + t) * CIN) * HW;
    #pragma unroll
    for (int ic = 0; ic < CIN; ++ic) {
        const float* ip = xin + (size_t)ic * HW;
        const float* wp = wi + ((size_t)oc * CIN + ic) * 25;
        for (int ky = 0; ky < 5; ++ky) {
            int yy = y + ky - 2;
            if (yy < 0 || yy >= Hn) continue;
            for (int kx = 0; kx < 5; ++kx) {
                int xx = x + kx - 2;
                if (xx < 0 || xx >= Wn) continue;
                sum = __builtin_fmaf(ip[yy * Wn + xx], wp[ky * 5 + kx], sum);
            }
        }
    }
    out[((size_t)b * HW + p) * FCH + oc] = sum;
}

// ---------------- h2f conv as 25 shifted WMMA GEMMs; f = tanh(fpre + conv) -> NHWC bf16 ----
// block: 4 image rows x 96 cols (384 px) of batch blockIdx.y; 8 waves, wave tile 48px x 32oc
// LDS: halo tile [8 rows][100 cols][64 ch] bf16 = 102,400 B
__global__ void k_h2f_wmma(const bf16_t* __restrict__ hbf,   // (B,HW,64) NHWC
                           const bf16_t* __restrict__ wt,    // [25][32][64]
                           const float*  __restrict__ fpre,  // (B,HW,32) NHWC
                           bf16_t* __restrict__ fnhwc) {     // (B,HW,32) NHWC
    extern __shared__ char smem_raw[];
    bf16_t* sH = (bf16_t*)smem_raw;
    const int tid = threadIdx.x;
    const int y0  = blockIdx.x * 4;
    const int b   = blockIdx.y;
    // stage 8x100x64 halo (zero-padded)
    for (int i = tid; i < 8 * 100 * 8; i += 256) {
        int cg = i & 7;
        int xx = (i >> 3) % 100;
        int yy = i / 800;
        int y = y0 + yy - 2, x = xx - 2;
        v8bf val = {};
        if (y >= 0 && y < Hn && x >= 0 && x < Wn)
            val = *(const v8bf*)(hbf + ((size_t)b * HW + y * Wn + x) * 64 + cg * 8);
        *(v8bf*)(sH + ((size_t)(yy * 100 + xx)) * 64 + cg * 8) = val;
    }
    __syncthreads();

    const int wave = tid >> 5, lane = tid & 31;
    const int half = lane >> 4, lmod = lane & 15;
    int rb[3];
    #pragma unroll
    for (int i = 0; i < 3; ++i) {
        int pix = (wave * 3 + i) * 16 + lmod;
        rb[i] = ((pix / Wn) * 100 + (pix % Wn)) * 64 + half * 8;   // sH element base
    }
    v8f acc[3][2] = {};
    for (int ky = 0; ky < 5; ++ky) {
        const bf16_t* wrow = wt + (size_t)(ky * 5) * 2048 + lmod * 64 + half * 8;
        int kyb0 = rb[0] + ky * 6400;
        int kyb1 = rb[1] + ky * 6400;
        int kyb2 = rb[2] + ky * 6400;
        #pragma unroll
        for (int kx = 0; kx < 5; ++kx) {
            #pragma unroll
            for (int s = 0; s < 2; ++s) {                 // K-slice (ic 0..31 / 32..63)
                frag16 b0, b1;
                const bf16_t* bp = wrow + kx * 2048 + s * 32;   // const offsets fold
                b0.h[0] = *(const v8bf*)(bp);
                b0.h[1] = *(const v8bf*)(bp + 16);
                b1.h[0] = *(const v8bf*)(bp + 16 * 64);
                b1.h[1] = *(const v8bf*)(bp + 16 * 64 + 16);
                const int cofs = kx * 64 + s * 32;              // const ds offset part
                #pragma unroll
                for (int i = 0; i < 3; ++i) {
                    int base = (i == 0) ? kyb0 : (i == 1) ? kyb1 : kyb2;
                    frag16 a;
                    a.h[0] = *(const v8bf*)(sH + base + cofs);
                    a.h[1] = *(const v8bf*)(sH + base + cofs + 16);
                    acc[i][0] = __builtin_amdgcn_wmma_f32_16x16x32_bf16(
                        false, a.v, false, b0.v, (short)0, acc[i][0], false, false);
                    acc[i][1] = __builtin_amdgcn_wmma_f32_16x16x32_bf16(
                        false, a.v, false, b1.v, (short)0, acc[i][1], false, false);
                }
            }
        }
    }
    // epilogue: f = tanh(fpre + acc); block rows are contiguous -> p = y0*96 + pix
    #pragma unroll
    for (int i = 0; i < 3; ++i) {
        int pixBase = (wave * 3 + i) * 16 + half * 8;
        #pragma unroll
        for (int nt = 0; nt < 2; ++nt) {
            int oc = nt * 16 + lmod;
            #pragma unroll
            for (int r = 0; r < 8; ++r) {
                size_t off = ((size_t)b * HW + (size_t)y0 * Wn + pixBase + r) * FCH + oc;
                fnhwc[off] = (bf16_t)tanhf(fpre[off] + acc[i][nt][r]);
            }
        }
    }
}

// ---------------- flows conv as 25 shifted WMMA GEMMs (N padded 26->32) ----------------
// LDS: halo tile [8][100][32] bf16 = 51,200 B
__global__ void k_flows_wmma(const bf16_t* __restrict__ fnhwc, // (B,HW,32)
                             const bf16_t* __restrict__ wt,    // [25][32][32]
                             const float*  __restrict__ bias,  // (26)
                             float* __restrict__ out) {        // (B,26,H,W)
    extern __shared__ char smem_raw[];
    bf16_t* sF = (bf16_t*)smem_raw;
    const int tid = threadIdx.x;
    const int y0  = blockIdx.x * 4;
    const int b   = blockIdx.y;
    for (int i = tid; i < 8 * 100 * 4; i += 256) {
        int cg = i & 3;
        int xx = (i >> 2) % 100;
        int yy = i / 400;
        int y = y0 + yy - 2, x = xx - 2;
        v8bf val = {};
        if (y >= 0 && y < Hn && x >= 0 && x < Wn)
            val = *(const v8bf*)(fnhwc + ((size_t)b * HW + y * Wn + x) * 32 + cg * 8);
        *(v8bf*)(sF + ((size_t)(yy * 100 + xx)) * 32 + cg * 8) = val;
    }
    __syncthreads();

    const int wave = tid >> 5, lane = tid & 31;
    const int half = lane >> 4, lmod = lane & 15;
    int rb[3];
    #pragma unroll
    for (int i = 0; i < 3; ++i) {
        int pix = (wave * 3 + i) * 16 + lmod;
        rb[i] = ((pix / Wn) * 100 + (pix % Wn)) * 32 + half * 8;
    }
    v8f acc[3][2] = {};
    for (int ky = 0; ky < 5; ++ky) {
        const bf16_t* wrow = wt + (size_t)(ky * 5) * 1024 + lmod * 32 + half * 8;
        int kyb0 = rb[0] + ky * 3200;
        int kyb1 = rb[1] + ky * 3200;
        int kyb2 = rb[2] + ky * 3200;
        #pragma unroll
        for (int kx = 0; kx < 5; ++kx) {
            frag16 b0, b1;
            const bf16_t* bp = wrow + kx * 1024;
            b0.h[0] = *(const v8bf*)(bp);
            b0.h[1] = *(const v8bf*)(bp + 16);
            b1.h[0] = *(const v8bf*)(bp + 16 * 32);
            b1.h[1] = *(const v8bf*)(bp + 16 * 32 + 16);
            const int cofs = kx * 32;
            #pragma unroll
            for (int i = 0; i < 3; ++i) {
                int base = (i == 0) ? kyb0 : (i == 1) ? kyb1 : kyb2;
                frag16 a;
                a.h[0] = *(const v8bf*)(sF + base + cofs);
                a.h[1] = *(const v8bf*)(sF + base + cofs + 16);
                acc[i][0] = __builtin_amdgcn_wmma_f32_16x16x32_bf16(
                    false, a.v, false, b0.v, (short)0, acc[i][0], false, false);
                acc[i][1] = __builtin_amdgcn_wmma_f32_16x16x32_bf16(
                    false, a.v, false, b1.v, (short)0, acc[i][1], false, false);
            }
        }
    }
    #pragma unroll
    for (int i = 0; i < 3; ++i) {
        int pixBase = (wave * 3 + i) * 16 + half * 8;
        #pragma unroll
        for (int nt = 0; nt < 2; ++nt) {
            int oc = nt * 16 + lmod;
            if (oc >= FL2) continue;
            float bv = bias[oc];
            #pragma unroll
            for (int r = 0; r < 8; ++r) {
                size_t p = (size_t)y0 * Wn + pixBase + r;
                out[((size_t)(b * FL2 + oc)) * HW + p] = acc[i][nt][r] + bv;
            }
        }
    }
}

// ---------------- grid-sample warp -> wrap GEMM A matrix [M=36864, K=832] bf16 ----------------
__global__ void k_warp(const float* __restrict__ flows, const float* __restrict__ h,
                       bf16_t* __restrict__ A) {
    int idx = blockIdx.x * blockDim.x + threadIdx.x;
    if (idx >= Bn * Ln * HW) return;
    int p = idx % HW;
    int l = (idx / HW) % Ln;
    int b = idx / (HW * Ln);
    int y = p / Wn, x = p % Wn;
    float fx = flows[((size_t)(b * FL2 + 2 * l)) * HW + p];
    float fy = flows[((size_t)(b * FL2 + 2 * l + 1)) * HW + p];
    float ix = (float)x - fx - 0.5f;
    float iy = (float)y - fy - 0.5f;
    float fx0 = floorf(ix), fy0 = floorf(iy);
    int ix0 = (int)fx0, iy0 = (int)fy0;
    int ix1 = ix0 + 1, iy1 = iy0 + 1;
    float wx1 = ix - fx0, wy1 = iy - fy0;
    float wx0 = 1.0f - wx1, wy0 = 1.0f - wy1;
    bool vx0 = (ix0 >= 0) & (ix0 < Wn), vx1 = (ix1 >= 0) & (ix1 < Wn);
    bool vy0 = (iy0 >= 0) & (iy0 < Hn), vy1 = (iy1 >= 0) & (iy1 < Hn);
    float w00 = (vy0 & vx0) ? wy0 * wx0 : 0.0f;
    float w01 = (vy0 & vx1) ? wy0 * wx1 : 0.0f;
    float w10 = (vy1 & vx0) ? wy1 * wx0 : 0.0f;
    float w11 = (vy1 & vx1) ? wy1 * wx1 : 0.0f;
    int cx0 = min(max(ix0, 0), Wn - 1), cx1 = min(max(ix1, 0), Wn - 1);
    int cy0 = min(max(iy0, 0), Hn - 1), cy1 = min(max(iy1, 0), Hn - 1);
    int o00 = cy0 * Wn + cx0, o01 = cy0 * Wn + cx1;
    int o10 = cy1 * Wn + cx0, o11 = cy1 * Wn + cx1;
    const float* hb = h + (size_t)b * HID * HW;
    bf16_t* arow = A + (size_t)(b * HW + p) * KK + l * HID;
    for (int c = 0; c < HID; ++c) {
        const float* hc = hb + (size_t)c * HW;
        float v = w00 * hc[o00] + w01 * hc[o01] + w10 * hc[o10] + w11 * hc[o11];
        arow[c] = (bf16_t)v;
    }
}

// ---------------- wrap 1x1 conv: WMMA GEMM h2h[M,192] = A[M,832] x W^T + bias ----------------
// Weight panel staged via CDNA5 async load (memory -> LDS, ASYNCcnt-tracked, no VGPR hop)
__global__ void k_gemm_wmma(const bf16_t* __restrict__ A, const bf16_t* __restrict__ Bt,
                            const float* __restrict__ bias, float* __restrict__ C) {
    extern __shared__ char smem_raw[];
    bf16_t* sB = (bf16_t*)smem_raw;                 // [64][832]
    const int tid   = threadIdx.x;
    const int nBase = blockIdx.y * 64;
    for (int i = tid; i < 64 * KK / 8; i += 256) {  // 6656 b128 chunks, 26/thread
        int n  = i / (KK / 8);
        int kc = (i % (KK / 8)) * 8;
        unsigned dst = (unsigned)(uintptr_t)(sB + (size_t)n * KK + kc);
        unsigned long long src =
            (unsigned long long)(uintptr_t)(Bt + (size_t)(nBase + n) * KK + kc);
        asm volatile("global_load_async_to_lds_b128 %0, %1, off"
                     :: "v"(dst), "v"(src) : "memory");
    }
    asm volatile("s_wait_asynccnt 0" ::: "memory");
    __syncthreads();

    const int wave = tid >> 5, lane = tid & 31;
    const int half = lane >> 4, lmod = lane & 15;
    const int mBase = blockIdx.x * 128 + wave * 16;
    const bf16_t* aRow = A + (size_t)(mBase + lmod) * KK;

    v8f acc0 = {}, acc1 = {}, acc2 = {}, acc3 = {};
    for (int k = 0; k < KK; k += 32) {
        frag16 a;
        a.h[0] = *(const v8bf*)(aRow + k + half * 8);
        a.h[1] = *(const v8bf*)(aRow + k + 16 + half * 8);
        #pragma unroll
        for (int nt = 0; nt < 4; ++nt) {
            frag16 bfr;
            const bf16_t* bp = sB + (size_t)(nt * 16 + lmod) * KK + k + half * 8;
            bfr.h[0] = *(const v8bf*)(bp);
            bfr.h[1] = *(const v8bf*)(bp + 16);
            v8f* pa = (nt == 0) ? &acc0 : (nt == 1) ? &acc1 : (nt == 2) ? &acc2 : &acc3;
            *pa = __builtin_amdgcn_wmma_f32_16x16x32_bf16(
                false, a.v, false, bfr.v, (short)0, *pa, false, false);
        }
    }
    v8f accs[4] = {acc0, acc1, acc2, acc3};
    #pragma unroll
    for (int nt = 0; nt < 4; ++nt) {
        int n = nBase + nt * 16 + lmod;
        float bv = bias[n];
        #pragma unroll
        for (int r = 0; r < 8; ++r) {
            int m = mBase + r + half * 8;
            C[(size_t)m * NN + n] = accs[nt][r] + bv;
        }
    }
}

// ---------------- fused gates: outputs[:,t], h (fp32 + NHWC bf16), h_last ----------------
__global__ void k_gates(const float* __restrict__ i2h_t, const float* __restrict__ h2h,
                        float* __restrict__ h, bf16_t* __restrict__ hbf,
                        float* __restrict__ outs, float* __restrict__ hlast, int t) {
    int idx = blockIdx.x * blockDim.x + threadIdx.x;
    if (idx >= Bn * HID * HW) return;
    int p = idx % HW;
    int c = (idx / HW) % HID;
    int b = idx / (HW * HID);
    size_t ib = ((size_t)b * NN) * HW + p;
    float r_t = i2h_t[ib + (size_t)c * HW];
    float u_t = i2h_t[ib + (size_t)(c + HID) * HW];
    float n_t = i2h_t[ib + (size_t)(c + 2 * HID) * HW];
    size_t row = ((size_t)b * HW + p) * NN;
    float r_h = h2h[row + c];
    float u_h = h2h[row + HID + c];
    float n_h = h2h[row + 2 * HID + c];
    float r = sigmoidf(r_t + r_h);
    float u = sigmoidf(u_t + u_h);
    float n = sigmoidf(n_t + r * n_h);
    size_t hoff = ((size_t)(b * HID + c)) * HW + p;
    float h_new = (1.0f - u) * n + u * h[hoff];
    h[hoff] = h_new;
    hbf[((size_t)b * HW + p) * HID + c] = (bf16_t)h_new;
    outs[((size_t)((b * Tn + t) * HID + c)) * HW + p] = h_new;
    if (hlast) hlast[hoff] = h_new;
}

// ---------------- host-side orchestration ----------------
extern "C" void kernel_launch(void* const* d_in, const int* in_sizes, int n_in,
                              void* d_out, int out_size, void* d_ws, size_t ws_size,
                              hipStream_t stream) {
    (void)in_sizes; (void)n_in; (void)out_size; (void)ws_size;
    const float* inputs  = (const float*)d_in[0];
    const float* state   = (const float*)d_in[1];
    const float* i2h_w   = (const float*)d_in[2];
    const float* i2h_b   = (const float*)d_in[3];
    const float* i2f_w   = (const float*)d_in[4];
    const float* i2f_b   = (const float*)d_in[5];
    const float* h2f_w   = (const float*)d_in[6];
    const float* h2f_b   = (const float*)d_in[7];
    const float* flows_w = (const float*)d_in[8];
    const float* flows_b = (const float*)d_in[9];
    const float* wrap_w  = (const float*)d_in[10];
    const float* wrap_b  = (const float*)d_in[11];

    char* ws = (char*)d_ws;
    float*  hbuf  = (float*)(ws);                       //  9,437,184
    bf16_t* hbf   = (bf16_t*)(ws + 9437184);            //  4,718,592 (B,HW,64)
    float*  i2h_t = (float*)(ws + 14155776);            // 28,311,552
    float*  fpre  = (float*)(ws + 42467328);            //  4,718,592 (B,HW,32)
    bf16_t* fnhwc = (bf16_t*)(ws + 47185920);           //  2,359,296 (B,HW,32)
    float*  flbuf = (float*)(ws + 49545216);            //  3,833,856 (B,26,HW)
    bf16_t* Abuf  = (bf16_t*)(ws + 53379072);           // 61,341,696
    float*  h2h   = (float*)(ws + 114720768);           // 28,311,552
    bf16_t* wb    = (bf16_t*)(ws + 143032320);          //    319,488
    bf16_t* whbf  = (bf16_t*)(ws + 143351808);          //    102,400 [25][32][64]
    bf16_t* wfbf  = (bf16_t*)(ws + 143454208);          //     51,200 [25][32][32]

    float* outs  = (float*)d_out;
    float* hlast = outs + (size_t)Bn * Tn * HID * HW;

    // --- one-time prep ---
    hipMemcpyAsync(hbuf, state, (size_t)Bn * HID * HW * sizeof(float),
                   hipMemcpyDeviceToDevice, stream);
    k_cvt_nhwc<<<(Bn * HID * HW + 255) / 256, 256, 0, stream>>>(state, hbf);
    k_cvt_bf16<<<(NN * KK + 255) / 256, 256, 0, stream>>>(wrap_w, wb, NN * KK);
    k_prep_h2f<<<(25 * 32 * 64 + 255) / 256, 256, 0, stream>>>(h2f_w, whbf);
    k_prep_flows<<<(25 * 32 * 32 + 255) / 256, 256, 0, stream>>>(flows_w, wfbf);

    const dim3 gemmGrid(MM / 128, NN / 64);
    const size_t gemmLds  = (size_t)64 * KK * sizeof(bf16_t);       // 106,496 B
    const dim3 convGrid(Hn / 4, Bn);
    const size_t h2fLds   = (size_t)8 * 100 * 64 * sizeof(bf16_t);  // 102,400 B
    const size_t flowsLds = (size_t)8 * 100 * 32 * sizeof(bf16_t);  //  51,200 B

    for (int t = 0; t < Tn; ++t) {
        k_conv3x3_i2h<<<(Bn * NN * HW + 255) / 256, 256, 0, stream>>>(
            inputs, i2h_w, i2h_b, i2h_t, t);
        k_i2f<<<(Bn * FCH * HW + 255) / 256, 256, 0, stream>>>(
            inputs, i2f_w, i2f_b, h2f_b, fpre, t);
        k_h2f_wmma<<<convGrid, 256, h2fLds, stream>>>(hbf, whbf, fpre, fnhwc);
        k_flows_wmma<<<convGrid, 256, flowsLds, stream>>>(fnhwc, wfbf, flows_b, flbuf);
        k_warp<<<(Bn * Ln * HW + 255) / 256, 256, 0, stream>>>(flbuf, hbuf, Abuf);
        k_gemm_wmma<<<gemmGrid, 256, gemmLds, stream>>>(Abuf, wb, wrap_b, h2h);
        k_gates<<<(Bn * HID * HW + 255) / 256, 256, 0, stream>>>(
            i2h_t, h2h, hbuf, hbf, outs, (t == Tn - 1) ? hlast : nullptr, t);
    }
}
